// STCADenseLayer_46316927320611
// MI455X (gfx1250) — matrix operand order repaired and checked by reading, not compiled, not run on hardware
//
#include <hip/hip_runtime.h>
#include <hip/hip_bf16.h>

typedef __attribute__((ext_vector_type(16))) __bf16 v16bf;
typedef __attribute__((ext_vector_type(8)))  __bf16 v8bf;
typedef __attribute__((ext_vector_type(8)))  float  v8f;

#define B_SZ   128
#define T_SZ   500
#define DIN    700
#define DOUT   512
#define KPAD   704          // DIN padded to multiple of 32
#define KC_X   22           // KPAD/32 k-chunks for x@w
#define KC_V   16           // DOUT/32 k-chunks for spk@v
#define NT     32           // DOUT/16 column tiles
#define M_ROWS (B_SZ*T_SZ)  // 64000
#define FRAG_H 512          // halfs per packed fragment (32 lanes * 16)

static __device__ __forceinline__ v16bf make_frag(v8bf lo, v8bf hi) {
  return __builtin_shufflevector(lo, hi, 0,1,2,3,4,5,6,7,8,9,10,11,12,13,14,15);
}

// ---------------- small prep kernels ----------------

__global__ void k_init_loss(float* loss) {
  if (threadIdx.x == 0 && blockIdx.x == 0) *loss = 0.f;
}

__global__ void k_norm(const float* __restrict__ w, float* __restrict__ nrm,
                       float* __restrict__ invn) {
  int d = blockIdx.x * blockDim.x + threadIdx.x;
  if (d >= DOUT) return;
  float s = 0.f;
  for (int c = 0; c < DIN; ++c) {
    float t = w[(size_t)c * DOUT + d];
    s += t * t;
  }
  nrm[d]  = s;
  invn[d] = 1.f / (s + 1e-8f);
}

// Pack w [DIN, DOUT] f32 -> bf16 B-fragment layout:
// frag(nt, kc): lane l holds col n = nt*16 + (l&15), K = kc*32 + (l>>4)*16 + j
__global__ void k_pack_w(const float* __restrict__ w, __bf16* __restrict__ wpk) {
  int tid = blockIdx.x * blockDim.x + threadIdx.x;
  if (tid >= KC_X * NT * 32) return;
  int lane = tid & 31;
  int kc   = (tid >> 5) % KC_X;
  int nt   = tid / (32 * KC_X);
  int n     = nt * 16 + (lane & 15);
  int kbase = kc * 32 + (lane >> 4) * 16;
  __bf16* dst = wpk + ((size_t)(nt * KC_X + kc) * 32 + lane) * 16;
  for (int j = 0; j < 16; ++j) {
    int k = kbase + j;
    float val = (k < DIN) ? w[(size_t)k * DOUT + n] : 0.f;
    dst[j] = (__bf16)val;
  }
}

__global__ void k_pack_v(const float* __restrict__ v, __bf16* __restrict__ vpk) {
  int tid = blockIdx.x * blockDim.x + threadIdx.x;
  if (tid >= KC_V * NT * 32) return;
  int lane = tid & 31;
  int kc   = (tid >> 5) % KC_V;
  int nt   = tid / (32 * KC_V);
  int n     = nt * 16 + (lane & 15);
  int kbase = kc * 32 + (lane >> 4) * 16;
  __bf16* dst = vpk + ((size_t)(nt * KC_V + kc) * 32 + lane) * 16;
  for (int j = 0; j < 16; ++j)
    dst[j] = (__bf16)v[(size_t)(kbase + j) * DOUT + n];
}

// x [M_ROWS, DIN] f32 -> xb [M_ROWS, KPAD] bf16 row-major, zero padded
__global__ void k_conv_x(const float* __restrict__ x, __bf16* __restrict__ xb) {
  size_t idx = (size_t)blockIdx.x * blockDim.x + threadIdx.x;
  if (idx >= (size_t)M_ROWS * KPAD) return;
  size_t m = idx / KPAD;
  int    k = (int)(idx % KPAD);
  xb[idx] = (k < DIN) ? (__bf16)x[m * DIN + k] : (__bf16)0.f;
}

// ---------------- h = x @ w  (WMMA bf16 -> f32) ----------------
// block = 256 threads (8 waves); block tile = 16 rows x 512 cols; wave tile = 16x64
// Constant-offset loads + manual 1-deep software pipeline (prefetch kc+1 while
// computing kc) so every load has >= 4 WMMAs of latency cover.

__global__ void __launch_bounds__(256)
k_gemm_h(const __bf16* __restrict__ xb, const __bf16* __restrict__ wpk,
         float* __restrict__ h) {
  int wave = threadIdx.x >> 5;
  int lane = threadIdx.x & 31;
  int row0 = blockIdx.x * 16;
  int lr   = (lane >> 4) * 8;

  const __bf16* arow = xb + (size_t)(row0 + (lane & 15)) * KPAD + lr * 2;
  const __bf16* wb = wpk + (size_t)(wave * 4) * KC_X * FRAG_H + (size_t)lane * 16;

  v8f acc[4] = {};

  // prologue: kc = 0
  v16bf a_cur = make_frag(*(const v8bf*)(arow), *(const v8bf*)(arow + 8));
  v16bf b_cur[4];
#pragma unroll
  for (int j = 0; j < 4; ++j)
    b_cur[j] = *(const v16bf*)(wb + (j * KC_X + 0) * FRAG_H);

#pragma unroll
  for (int kc = 0; kc < KC_X; ++kc) {
    v16bf a_nxt = a_cur;
    v16bf b_nxt[4] = {b_cur[0], b_cur[1], b_cur[2], b_cur[3]};
    if (kc + 1 < KC_X) {
      a_nxt = make_frag(*(const v8bf*)(arow + (kc + 1) * 32),
                        *(const v8bf*)(arow + (kc + 1) * 32 + 8));
#pragma unroll
      for (int j = 0; j < 4; ++j)
        b_nxt[j] = *(const v16bf*)(wb + (j * KC_X + kc + 1) * FRAG_H);
    }
    acc[0] = __builtin_amdgcn_wmma_f32_16x16x32_bf16(false, a_cur, false, b_cur[0], (short)0, acc[0], false, false);
    acc[1] = __builtin_amdgcn_wmma_f32_16x16x32_bf16(false, a_cur, false, b_cur[1], (short)0, acc[1], false, false);
    acc[2] = __builtin_amdgcn_wmma_f32_16x16x32_bf16(false, a_cur, false, b_cur[2], (short)0, acc[2], false, false);
    acc[3] = __builtin_amdgcn_wmma_f32_16x16x32_bf16(false, a_cur, false, b_cur[3], (short)0, acc[3], false, false);
    a_cur = a_nxt;
    b_cur[0] = b_nxt[0]; b_cur[1] = b_nxt[1];
    b_cur[2] = b_nxt[2]; b_cur[3] = b_nxt[3];
  }

  float* hout = h + (size_t)(row0 + lr) * DOUT + wave * 64 + (lane & 15);
#pragma unroll
  for (int nt2 = 0; nt2 < 4; ++nt2)
#pragma unroll
    for (int i = 0; i < 8; ++i)
      hout[(size_t)i * DOUT + nt2 * 16] = acc[nt2][i];
}

// ---------------- recurrent scan ----------------
// 8 blocks x 512 threads (16 waves). Block owns 16 batch rows, runs all T steps.
// Wave owns cols [wave*32, wave*32+32); spike matrix ping-pongs through LDS (bf16).
// h loads hoisted to step start; v/spike fragments software-pipelined 1 deep.

__global__ void __launch_bounds__(512)
k_recurrent(const float* __restrict__ h, const __bf16* __restrict__ vpk,
            const float* __restrict__ nrm, const float* __restrict__ invn,
            const float* __restrict__ bb, const float* __restrict__ dmp,
            const float* __restrict__ dsp, float* __restrict__ spk_rec,
            float* __restrict__ loss) {
  __shared__ __bf16 spk_lds[2][16][DOUT];   // 32 KB

  int wave = threadIdx.x >> 5;
  int lane = threadIdx.x & 31;
  int r0   = blockIdx.x * 16;
  int lr   = (lane >> 4) * 8;               // local row base (C/D layout)

  for (int i = threadIdx.x; i < 16 * DOUT; i += blockDim.x)
    ((__bf16*)spk_lds[0])[i] = (__bf16)0.f;
  __syncthreads();

  float dm = dmp[0], dsc = dsp[0];
  float nv[2], iv[2], bv[2];
#pragma unroll
  for (int tt = 0; tt < 2; ++tt) {
    int col = wave * 32 + tt * 16 + (lane & 15);
    nv[tt] = nrm[col]; iv[tt] = invn[col]; bv[tt] = bb[col];
  }

  size_t base0 = (size_t)(r0 + lr) * T_SZ * DOUT + wave * 32 + (lane & 15);
  const float* hp = h + base0;
  float*       sp = spk_rec + base0;
  const __bf16* vb = vpk + (size_t)(wave * 2) * KC_V * FRAG_H + (size_t)lane * 16;

  float E[2][8] = {}, Mst[2][8] = {}, Sst[2][8] = {}, spko[2][8] = {};
  float lsum = 0.f;
  int buf = 0;

#pragma unroll 1
  for (int t = 0; t < T_SZ; ++t) {
    // hoist h loads: issue before the WMMA chain (32 WMMAs of latency cover)
    float hin[2][8];
#pragma unroll
    for (int tt = 0; tt < 2; ++tt)
#pragma unroll
      for (int i = 0; i < 8; ++i)
        hin[tt][i] = hp[(size_t)i * (T_SZ * DOUT) + tt * 16];

    // acc = spk_old @ v ; 1-deep pipelined fragment loads
    v8f acc[2] = {};
    const __bf16* ar = &spk_lds[buf][lane & 15][lr * 2];

    v16bf a_cur = make_frag(*(const v8bf*)(ar), *(const v8bf*)(ar + 8));
    v16bf b_cur[2];
    b_cur[0] = *(const v16bf*)(vb + (0 * KC_V + 0) * FRAG_H);
    b_cur[1] = *(const v16bf*)(vb + (1 * KC_V + 0) * FRAG_H);

#pragma unroll
    for (int kc = 0; kc < KC_V; ++kc) {
      v16bf a_nxt = a_cur;
      v16bf b_nxt[2] = {b_cur[0], b_cur[1]};
      if (kc + 1 < KC_V) {
        a_nxt = make_frag(*(const v8bf*)(ar + (kc + 1) * 32),
                          *(const v8bf*)(ar + (kc + 1) * 32 + 8));
        b_nxt[0] = *(const v16bf*)(vb + (0 * KC_V + kc + 1) * FRAG_H);
        b_nxt[1] = *(const v16bf*)(vb + (1 * KC_V + kc + 1) * FRAG_H);
      }
      acc[0] = __builtin_amdgcn_wmma_f32_16x16x32_bf16(false, a_cur, false, b_cur[0], (short)0, acc[0], false, false);
      acc[1] = __builtin_amdgcn_wmma_f32_16x16x32_bf16(false, a_cur, false, b_cur[1], (short)0, acc[1], false, false);
      a_cur = a_nxt;
      b_cur[0] = b_nxt[0]; b_cur[1] = b_nxt[1];
    }

    // state update + threshold
#pragma unroll
    for (int tt = 0; tt < 2; ++tt) {
      int col = wave * 32 + tt * 16 + (lane & 15);
#pragma unroll
      for (int i = 0; i < 8; ++i) {
        float inp  = hin[tt][i] + acc[tt][i];
        E[tt][i]   = dm  * nv[tt] * (E[tt][i] + spko[tt][i]);
        Mst[tt][i] = dm  * (Mst[tt][i] + inp);
        Sst[tt][i] = dsc * (Sst[tt][i] + inp);
        float mthr = (Mst[tt][i] - Sst[tt][i] - E[tt][i]) * iv[tt] - bv[tt];
        float ns = (mthr > 0.f) ? 1.f : 0.f;
        sp[(size_t)i * (T_SZ * DOUT) + tt * 16] = ns;
        lsum += ns;
        spko[tt][i] = ns;
        spk_lds[buf ^ 1][lr + i][col] = (__bf16)ns;
      }
    }
    __syncthreads();
    buf ^= 1;
    hp += DOUT;
    sp += DOUT;
  }
  atomicAdd(loss, lsum * (0.5f / (float)((size_t)B_SZ * T_SZ * DOUT)));
}

// ---------------- launcher ----------------

extern "C" void kernel_launch(void* const* d_in, const int* in_sizes, int n_in,
                              void* d_out, int out_size, void* d_ws, size_t ws_size,
                              hipStream_t stream) {
  const float* x   = (const float*)d_in[0];
  const float* w   = (const float*)d_in[1];
  const float* v   = (const float*)d_in[2];
  const float* b   = (const float*)d_in[3];
  const float* dmv = (const float*)d_in[4];
  const float* dsv = (const float*)d_in[5];

  float* spk_rec = (float*)d_out;
  float* loss    = spk_rec + (size_t)B_SZ * T_SZ * DOUT;

  char* ws = (char*)d_ws;
  size_t off = 0;
  float*  nrm  = (float*)(ws + off);  off += DOUT * sizeof(float);
  float*  invn = (float*)(ws + off);  off += DOUT * sizeof(float);
  __bf16* wpk  = (__bf16*)(ws + off); off += (size_t)KC_X * NT * FRAG_H * 2;
  __bf16* vpk  = (__bf16*)(ws + off); off += (size_t)KC_V * NT * FRAG_H * 2;
  __bf16* xb   = (__bf16*)(ws + off); off += (size_t)M_ROWS * KPAD * 2;
  float*  h    = (float*)(ws + off);  // M_ROWS*DOUT*4

  k_init_loss<<<1, 1, 0, stream>>>(loss);
  k_norm<<<(DOUT + 255) / 256, 256, 0, stream>>>(w, nrm, invn);
  k_pack_w<<<(KC_X * NT * 32 + 255) / 256, 256, 0, stream>>>(w, wpk);
  k_pack_v<<<(KC_V * NT * 32 + 255) / 256, 256, 0, stream>>>(v, vpk);
  size_t xtotal = (size_t)M_ROWS * KPAD;
  k_conv_x<<<(int)((xtotal + 255) / 256), 256, 0, stream>>>(x, xb);
  k_gemm_h<<<M_ROWS / 16, 256, 0, stream>>>(xb, wpk, h);
  k_recurrent<<<B_SZ / 16, 512, 0, stream>>>(h, vpk, nrm, invn, b, dmv, dsv,
                                             spk_rec, loss);
}
